// MultiHeadSelfAttention_23605140259528
// MI455X (gfx1250) — matrix operand order repaired and checked by reading, not compiled, not run on hardware
//
#include <hip/hip_runtime.h>

// MHSA forward for MI455X (gfx1250, wave32, WMMA).
// B=2, S=2048, D=1024, H=16, dh=64. fp16 WMMA (16x16x32) with fp32 accum.
// GEMMs and attention are software-pipelined: next-iteration fragments are
// loaded into distinct registers before the current WMMAs consume theirs,
// so s_wait_loadcnt tolerates in-flight loads instead of draining to 0.

typedef __attribute__((ext_vector_type(16))) _Float16 v16h;
typedef __attribute__((ext_vector_type(8)))  _Float16 v8h;
typedef __attribute__((ext_vector_type(8)))  float    v8f;

#define B_   2
#define S_   2048
#define D_   1024
#define H_   16
#define DH_  64
#define N3_  3072
#define M_   4096   // B_*S_
#define NEG_BIG (-1e30f)

__device__ __forceinline__ v8f wmma16(v16h a, v16h b, v8f c) {
  return __builtin_amdgcn_wmma_f32_16x16x32_f16(false, a, false, b,
                                                (short)0, c, false, false);
}

// A-fragment (16x32 f16, row-major source, leading dim ld):
// lane<16: row=lane, K={0..7,16..23}; lane>=16: row=lane-16, K={8..15,24..31}
__device__ __forceinline__ v16h load_a_frag(const _Float16* base, int ld) {
  int lane = threadIdx.x & 31;
  const _Float16* p = base + (lane & 15) * ld + ((lane >> 4) << 3);
  v8h lo = *(const v8h*)p;
  v8h hi = *(const v8h*)(p + 16);
  return __builtin_shufflevector(lo, hi, 0,1,2,3,4,5,6,7,8,9,10,11,12,13,14,15);
}

// B-fragment (32x16 f16). baseT is row-major B^T (row = output col n):
// lanes 0-15 hold K={0..7,16..23}, lanes 16-31 hold K={8..15,24..31}.
__device__ __forceinline__ v16h load_b_frag(const _Float16* baseT, int ld) {
  int lane = threadIdx.x & 31;
  const _Float16* p = baseT + (lane & 15) * ld + ((lane >> 4) << 3);
  v8h lo = *(const v8h*)p;
  v8h hi = *(const v8h*)(p + 16);
  return __builtin_shufflevector(lo, hi, 0,1,2,3,4,5,6,7,8,9,10,11,12,13,14,15);
}

// ---------------- conversion kernels ----------------

__global__ void cvt_f32_f16(const float* __restrict__ src,
                            _Float16* __restrict__ dst, int n) {
  int i = blockIdx.x * blockDim.x + threadIdx.x;
  if (i < n) dst[i] = (_Float16)src[i];
}

// src [R][C] f32 row-major -> dst [C][R] f16 (transposed)
__global__ void cvt_t_f16(const float* __restrict__ src,
                          _Float16* __restrict__ dst, int R, int C) {
  int i = blockIdx.x * blockDim.x + threadIdx.x;
  if (i < R * C) {
    int c = i / R, r = i - c * R;
    dst[i] = (_Float16)src[r * C + c];
  }
}

// ---------------- QKV projection ----------------
// Wave owns a 32x64 tile (2 A-frags x 4 B-frags, 8 accumulators), with
// explicit double-buffered fragments for load/WMMA overlap.
// q,k stored [B,H,S,dh] row-major; v stored transposed [B,H,dh,S].
__global__ __launch_bounds__(256) void gemm_qkv(
    const _Float16* __restrict__ xh, const _Float16* __restrict__ wt,
    _Float16* __restrict__ qh, _Float16* __restrict__ kh,
    _Float16* __restrict__ vt) {
  int wid  = blockIdx.x * 8 + (threadIdx.x >> 5);
  int lane = threadIdx.x & 31;
  int half = lane >> 4, lcol = lane & 15;
  int mt = wid / 48, nt = wid - mt * 48;
  int m0 = mt * 32, n0 = nt * 64;

  const _Float16* A0 = xh + m0 * D_;
  const _Float16* A1 = xh + (m0 + 16) * D_;
  const _Float16* Bt = wt + n0 * D_;

  v8f acc0[4] = {}, acc1[4] = {};
  v16h a0 = load_a_frag(A0, D_);
  v16h a1 = load_a_frag(A1, D_);
  v16h b0 = load_b_frag(Bt + 0 * D_, D_);
  v16h b1 = load_b_frag(Bt + 16 * D_, D_);
  v16h b2 = load_b_frag(Bt + 32 * D_, D_);
  v16h b3 = load_b_frag(Bt + 48 * D_, D_);

  for (int k0 = 0; k0 < D_; k0 += 32) {
    int kn = (k0 + 32 < D_) ? k0 + 32 : k0;   // clamped prefetch (branchless)
    v16h na0 = load_a_frag(A0 + kn, D_);
    v16h na1 = load_a_frag(A1 + kn, D_);
    v16h nb0 = load_b_frag(Bt + 0 * D_ + kn, D_);
    v16h nb1 = load_b_frag(Bt + 16 * D_ + kn, D_);
    v16h nb2 = load_b_frag(Bt + 32 * D_ + kn, D_);
    v16h nb3 = load_b_frag(Bt + 48 * D_ + kn, D_);

    acc0[0] = wmma16(a0, b0, acc0[0]);
    acc1[0] = wmma16(a1, b0, acc1[0]);
    acc0[1] = wmma16(a0, b1, acc0[1]);
    acc1[1] = wmma16(a1, b1, acc1[1]);
    acc0[2] = wmma16(a0, b2, acc0[2]);
    acc1[2] = wmma16(a1, b2, acc1[2]);
    acc0[3] = wmma16(a0, b3, acc0[3]);
    acc1[3] = wmma16(a1, b3, acc1[3]);

    a0 = na0; a1 = na1; b0 = nb0; b1 = nb1; b2 = nb2; b3 = nb3;
  }

  int which = n0 >> 10;           // 0=q 1=k 2=v
  int h = (n0 & 1023) >> 6;       // 64-wide strips never straddle heads
#pragma unroll
  for (int g = 0; g < 2; ++g) {
    v8f* acc = g ? acc1 : acc0;
#pragma unroll
    for (int i = 0; i < 4; ++i) {
#pragma unroll
      for (int j = 0; j < 8; ++j) {
        int m = m0 + g * 16 + j + 8 * half;
        int b = m >> 11, s = m & 2047;
        int dd = i * 16 + lcol;
        int bh = b * H_ + h;
        _Float16 val = (_Float16)acc[i][j];
        if (which == 0)      qh[(bh * S_ + s) * DH_ + dd] = val;
        else if (which == 1) kh[(bh * S_ + s) * DH_ + dd] = val;
        else                 vt[(bh * DH_ + dd) * S_ + s] = val;
      }
    }
  }
}

// ---------------- flash attention ----------------
// One wave = one 16-query tile of one (b,h). Key blocks of 32, online
// softmax. Next block's K-frags are preloaded and this block's V-frags are
// issued before the softmax VALU section so both WMMA stages find data ready.
__global__ __launch_bounds__(256) void attn_kernel(
    const _Float16* __restrict__ qh, const _Float16* __restrict__ kh,
    const _Float16* __restrict__ vt, _Float16* __restrict__ outh) {
  __shared__ float ldsP[8][16 * 32];   // per-wave C->A transpose slab
  int wave = blockIdx.x * 8 + (threadIdx.x >> 5);
  int lane = threadIdx.x & 31;
  int half = lane >> 4, lcol = lane & 15;
  int qt = wave & 127;        // S_/16 query tiles
  int bh = wave >> 7;         // b*H + h
  int q0 = qt * 16;
  const _Float16* Qb = qh + (bh * S_ + q0) * DH_;
  const _Float16* Kb = kh + bh * S_ * DH_;
  const _Float16* Vb = vt + bh * DH_ * S_;
  float* myP = ldsP[threadIdx.x >> 5];

  v16h qA0 = load_a_frag(Qb, DH_);        // d 0..31
  v16h qA1 = load_a_frag(Qb + 32, DH_);   // d 32..63

  float mrow[8], srow[8];
#pragma unroll
  for (int j = 0; j < 8; ++j) { mrow[j] = NEG_BIG; srow[j] = 0.f; }
  v8f o[4] = {};

  int kend = q0 + 16;                 // keys < kend participate
  // prologue: K-frags for block kb=0 (clamp 2nd tile if inactive)
  int o1p = (16 < kend) ? 16 * DH_ : 0;
  v16h kf00 = load_b_frag(Kb, DH_);
  v16h kf01 = load_b_frag(Kb + 32, DH_);
  v16h kf10 = load_b_frag(Kb + o1p, DH_);
  v16h kf11 = load_b_frag(Kb + o1p + 32, DH_);

  for (int kb = 0; kb < kend; kb += 32) {
    bool act1 = (kb + 16) < kend;     // uniform per wave

    // --- two 16x16 score tiles ---
    v8f c0 = {}, c1 = {};
    c0 = wmma16(qA0, kf00, c0);
    c0 = wmma16(qA1, kf01, c0);
    c1 = wmma16(qA0, kf10, c1);
    c1 = wmma16(qA1, kf11, c1);

    // --- preload next block's K-frags (clamped, branchless) ---
    int kn = (kb + 32 < kend) ? kb + 32 : kb;
    bool act1n = (kn + 16) < kend;
    const _Float16* Ktn = Kb + kn * DH_;
    int o1 = act1n ? 16 * DH_ : 0;
    kf00 = load_b_frag(Ktn, DH_);
    kf01 = load_b_frag(Ktn + 32, DH_);
    kf10 = load_b_frag(Ktn + o1, DH_);
    kf11 = load_b_frag(Ktn + o1 + 32, DH_);

    // --- issue this block's V-frags early (hide behind softmax VALU) ---
    v16h vf0 = load_b_frag(Vb + 0  * S_ + kb, S_);
    v16h vf1 = load_b_frag(Vb + 16 * S_ + kb, S_);
    v16h vf2 = load_b_frag(Vb + 32 * S_ + kb, S_);
    v16h vf3 = load_b_frag(Vb + 48 * S_ + kb, S_);

    // --- mask + scale ---
    float sc0[8], sc1[8];
#pragma unroll
    for (int j = 0; j < 8; ++j) {
      int qg = q0 + j + 8 * half;
      int kg = kb + lcol;
      float x0 = c0[j] * 0.125f;          // 1/sqrt(64)
      float x1 = c1[j] * 0.125f;
      sc0[j] = (kg > qg) ? NEG_BIG : x0;
      sc1[j] = (act1 && (kg + 16) <= qg) ? x1 : ((kg + 16) <= qg ? x1 : NEG_BIG);
      sc1[j] = (!act1 || (kg + 16) > qg) ? NEG_BIG : x1;
    }

    // --- online softmax (row reductions stay within 16-lane halves) ---
    float rscale[8];
#pragma unroll
    for (int j = 0; j < 8; ++j) {
      float tmax = fmaxf(sc0[j], sc1[j]);
      tmax = fmaxf(tmax, __shfl_xor(tmax, 1));
      tmax = fmaxf(tmax, __shfl_xor(tmax, 2));
      tmax = fmaxf(tmax, __shfl_xor(tmax, 4));
      tmax = fmaxf(tmax, __shfl_xor(tmax, 8));
      float mnew = fmaxf(mrow[j], tmax);
      float p0 = __expf(sc0[j] - mnew);
      float p1 = __expf(sc1[j] - mnew);
      float rsum = p0 + p1;
      rsum += __shfl_xor(rsum, 1);
      rsum += __shfl_xor(rsum, 2);
      rsum += __shfl_xor(rsum, 4);
      rsum += __shfl_xor(rsum, 8);
      float so = __expf(mrow[j] - mnew);
      srow[j] = srow[j] * so + rsum;
      mrow[j] = mnew;
      rscale[j] = so;
      int row = j + 8 * half;
      myP[row * 32 + lcol] = p0;
      myP[row * 32 + 16 + lcol] = p1;
    }
#pragma unroll
    for (int d = 0; d < 4; ++d)
#pragma unroll
      for (int j = 0; j < 8; ++j) o[d][j] *= rscale[j];

    // --- read P back as 16x32 A-frag (in-wave DS ops are in-order) ---
    v16h pA;
    {
      int row = lcol;
      int kb8 = half << 3;
#pragma unroll
      for (int j = 0; j < 8; ++j)
        pA[j] = (_Float16)myP[row * 32 + kb8 + j];
#pragma unroll
      for (int j = 0; j < 8; ++j)
        pA[8 + j] = (_Float16)myP[row * 32 + 16 + kb8 + j];
    }

    // --- O += P @ V ---
    o[0] = wmma16(pA, vf0, o[0]);
    o[1] = wmma16(pA, vf1, o[1]);
    o[2] = wmma16(pA, vf2, o[2]);
    o[3] = wmma16(pA, vf3, o[3]);
  }

  // --- normalize + store [B,S,D] f16 ---
  int b = bh >> 4, h = bh & 15;
  _Float16* Ob = outh + (b * S_ + q0) * D_ + h * DH_;
#pragma unroll
  for (int j = 0; j < 8; ++j) {
    float inv = 1.0f / srow[j];
    int row = j + 8 * half;
#pragma unroll
    for (int d = 0; d < 4; ++d)
      Ob[row * D_ + d * 16 + lcol] = (_Float16)(o[d][j] * inv);
  }
}

// ---------------- output projection (fp32 result) ----------------
__global__ __launch_bounds__(256) void gemm_out(
    const _Float16* __restrict__ ah, const _Float16* __restrict__ wt,
    float* __restrict__ out) {
  int wid  = blockIdx.x * 8 + (threadIdx.x >> 5);
  int lane = threadIdx.x & 31;
  int half = lane >> 4, lcol = lane & 15;
  int mt = wid >> 4, nt = wid & 15;
  int m0 = mt * 32, n0 = nt * 64;

  const _Float16* A0 = ah + m0 * D_;
  const _Float16* A1 = ah + (m0 + 16) * D_;
  const _Float16* Bt = wt + n0 * D_;

  v8f acc0[4] = {}, acc1[4] = {};
  v16h a0 = load_a_frag(A0, D_);
  v16h a1 = load_a_frag(A1, D_);
  v16h b0 = load_b_frag(Bt + 0 * D_, D_);
  v16h b1 = load_b_frag(Bt + 16 * D_, D_);
  v16h b2 = load_b_frag(Bt + 32 * D_, D_);
  v16h b3 = load_b_frag(Bt + 48 * D_, D_);

  for (int k0 = 0; k0 < D_; k0 += 32) {
    int kn = (k0 + 32 < D_) ? k0 + 32 : k0;
    v16h na0 = load_a_frag(A0 + kn, D_);
    v16h na1 = load_a_frag(A1 + kn, D_);
    v16h nb0 = load_b_frag(Bt + 0 * D_ + kn, D_);
    v16h nb1 = load_b_frag(Bt + 16 * D_ + kn, D_);
    v16h nb2 = load_b_frag(Bt + 32 * D_ + kn, D_);
    v16h nb3 = load_b_frag(Bt + 48 * D_ + kn, D_);

    acc0[0] = wmma16(a0, b0, acc0[0]);
    acc1[0] = wmma16(a1, b0, acc1[0]);
    acc0[1] = wmma16(a0, b1, acc0[1]);
    acc1[1] = wmma16(a1, b1, acc1[1]);
    acc0[2] = wmma16(a0, b2, acc0[2]);
    acc1[2] = wmma16(a1, b2, acc1[2]);
    acc0[3] = wmma16(a0, b3, acc0[3]);
    acc1[3] = wmma16(a1, b3, acc1[3]);

    a0 = na0; a1 = na1; b0 = nb0; b1 = nb1; b2 = nb2; b3 = nb3;
  }

#pragma unroll
  for (int g = 0; g < 2; ++g) {
    v8f* acc = g ? acc1 : acc0;
#pragma unroll
    for (int i = 0; i < 4; ++i)
#pragma unroll
      for (int j = 0; j < 8; ++j) {
        int m = m0 + g * 16 + j + 8 * half;
        out[m * D_ + n0 + i * 16 + lcol] = acc[i][j];
      }
  }
}

// ---------------- launch ----------------
extern "C" void kernel_launch(void* const* d_in, const int* in_sizes, int n_in,
                              void* d_out, int out_size, void* d_ws,
                              size_t ws_size, hipStream_t stream) {
  const float* x     = (const float*)d_in[0];
  // d_in[1] = causal mask (int32) — causality applied analytically.
  const float* w_qkv = (const float*)d_in[2];
  const float* w_out = (const float*)d_in[3];
  float* out = (float*)d_out;

  char* ws = (char*)d_ws;
  _Float16* xh    = (_Float16*)ws;  ws += (size_t)M_ * D_ * 2;
  _Float16* wqkvt = (_Float16*)ws;  ws += (size_t)N3_ * D_ * 2;
  _Float16* woutt = (_Float16*)ws;  ws += (size_t)D_ * D_ * 2;
  _Float16* qh    = (_Float16*)ws;  ws += (size_t)M_ * D_ * 2;
  _Float16* kh    = (_Float16*)ws;  ws += (size_t)M_ * D_ * 2;
  _Float16* vt    = (_Float16*)ws;  ws += (size_t)M_ * D_ * 2;
  _Float16* aoh   = (_Float16*)ws;

  cvt_f32_f16<<<(M_ * D_) / 256, 256, 0, stream>>>(x, xh, M_ * D_);
  cvt_t_f16<<<(N3_ * D_) / 256, 256, 0, stream>>>(w_qkv, wqkvt, D_, N3_);
  cvt_t_f16<<<(D_ * D_) / 256, 256, 0, stream>>>(w_out, woutt, D_, D_);

  // 128 m-tiles(32) * 48 n-strips(64) = 6144 waves / 8 per block
  gemm_qkv<<<768, 256, 0, stream>>>(xh, wqkvt, qh, kh, vt);

  // 32 (b,h) * 128 q-tiles = 4096 waves
  attn_kernel<<<512, 256, 0, stream>>>(qh, kh, vt, aoh);

  // 128 m-tiles(32) * 16 n-strips(64) = 2048 waves
  gemm_out<<<256, 256, 0, stream>>>(aoh, woutt, out);
}
